// _RPN_5385888989435
// MI455X (gfx1250) — matrix operand order repaired and checked by reading, not compiled, not run on hardware
//
#include <hip/hip_runtime.h>
#include <stdint.h>

// ---------------- problem constants ----------------
#define B_    8
#define C_IN  1024
#define C_MID 512
#define H_    64
#define W_    48
#define HW    3072          // H_*W_
#define M_TOT 24576         // B_*H_*W_
#define NH    64            // padded head cols (18 cls + 36 bbox + 10 pad)
#define HP    66            // padded height
#define WP    50            // padded width

#define OFF_BBOX 442368     // 8*18*64*48
#define OFF_LCLS 1327104    // + 8*36*64*48

typedef __attribute__((ext_vector_type(16))) __bf16 v16bf;
typedef __attribute__((ext_vector_type(8)))  float  v8f;
typedef __attribute__((ext_vector_type(4)))  unsigned int v4u;
typedef __attribute__((ext_vector_type(8)))  int    v8i_;
typedef __attribute__((ext_vector_type(4)))  int    v4i_;

#if defined(__has_builtin)
#if __has_builtin(__builtin_amdgcn_tensor_load_to_lds)
#define HAVE_TDM 1
#endif
#endif

union FragBF { uint4 q[2]; v16bf v; };

// pack high 16 bits of two f32 -> one dword of two bf16 (truncate), 1 VALU op
__device__ __forceinline__ unsigned int pk_bf(unsigned int lo, unsigned int hi) {
  return __builtin_amdgcn_perm(hi, lo, 0x07060302u);
}
__device__ __forceinline__ unsigned short f2bf(float f) {
  return (unsigned short)(__float_as_uint(f) >> 16);
}

// ---------------- input pre-pass: NCHW f32 -> padded NHWC bf16 ----------------
__global__ void zero_xpad(uint4* __restrict__ p, int n4) {
  int i = blockIdx.x * blockDim.x + threadIdx.x;
  if (i < n4) p[i] = make_uint4(0u, 0u, 0u, 0u);
}

// grid (24 pix-blocks, 32 c-blocks, 8 batch), block 256
__global__ __launch_bounds__(256)
void cvt_nhwc(const float* __restrict__ in, unsigned short* __restrict__ out) {
  __shared__ __align__(16) unsigned short T[128 * 32];
  const int t  = threadIdx.x;
  const int p0 = blockIdx.x * 128;
  const int c0 = blockIdx.y * 32;
  const int b  = blockIdx.z;
  const int pix = t & 127;
  const int cb  = (t >> 7) * 16;
  const float* src = in + ((size_t)(b * C_IN + c0 + cb) * HW + p0 + pix);
  unsigned int u[16];
#pragma unroll
  for (int i = 0; i < 16; ++i)
    u[i] = __float_as_uint(src[(size_t)i * HW]);
#pragma unroll
  for (int i = 0; i < 8; ++i)
    *(unsigned int*)&T[pix * 32 + cb + 2 * i] = pk_bf(u[2 * i], u[2 * i + 1]);
  __syncthreads();
  const int wp_ = t >> 1, ch = (t & 1) * 16;
  const int pp = p0 + wp_;
  const int y = pp / W_, x = pp % W_;
  unsigned short* dst = out + ((size_t)((b * HP + y + 1) * WP) + x + 1) * C_IN + c0 + ch;
  const uint4* s4 = (const uint4*)&T[wp_ * 32 + ch];
  ((uint4*)dst)[0] = s4[0];
  ((uint4*)dst)[1] = s4[1];
}

// ---------------- weight packing ----------------
__global__ void pack_conv_w(const float* __restrict__ w, unsigned short* __restrict__ wp) {
  int i = blockIdx.x * blockDim.x + threadIdx.x;
  if (i >= C_MID * C_IN * 9) return;
  int t = i % 9;
  int c = (i / 9) % C_IN;
  int o = i / (9 * C_IN);
  wp[((size_t)t * C_MID + o) * C_IN + c] = f2bf(w[i]);
}

__global__ void pack_heads(const float* __restrict__ wc, const float* __restrict__ bc,
                           const float* __restrict__ wb, const float* __restrict__ bb,
                           unsigned short* __restrict__ wh, float* __restrict__ bh) {
  int i = blockIdx.x * blockDim.x + threadIdx.x;
  if (i >= NH * C_MID) return;
  int n = i >> 9, k = i & 511;
  float v = 0.f;
  if (n < 18)       v = wc[n * C_MID + k];
  else if (n < 54)  v = wb[(n - 18) * C_MID + k];
  wh[i] = f2bf(v);
  if (k == 0) {
    float bv = 0.f;
    if (n < 18)      bv = bc[n];
    else if (n < 54) bv = bb[n - 18];
    bh[n] = bv;
  }
}

__global__ void zero_accum(float* a) { if (threadIdx.x < 4) a[threadIdx.x] = 0.f; }

// ---------------- 3x3 conv + ReLU, implicit-GEMM bf16 WMMA ----------------
// tile 256(M) x 128(N), 512 threads = 16 wave32, grid (96, 4)
// A tiles: vector b128 path.  B tiles: Tensor Data Mover -> LDS (pad feature
// reproduces the stride-40-halfword layout: 64B row + 16B pad = 80B).
__global__ __launch_bounds__(512)
void conv3x3_relu_wmma(const unsigned short* __restrict__ xcl,
                       const unsigned short* __restrict__ wp,
                       const float* __restrict__ bias,
                       unsigned short* __restrict__ act) {
  __shared__ __align__(16) unsigned short As[2][256 * 40];
  __shared__ __align__(16) unsigned short Bs[2][128 * 40];

  const int tid  = threadIdx.x;
  const int lane = tid & 31;
  const int wave = tid >> 5;              // 0..15
  const int wm   = wave & 7;              // 8 M quadrants (32 rows)
  const int wn   = wave >> 3;             // 2 N halves (64 cols)
  const int m0   = blockIdx.x * 256;
  const int n0   = blockIdx.y * 128;

  // loop-invariant A addressing: 2 b128 chunks / thread
  const unsigned short* aptr[2];
  int sA[2];
#pragma unroll
  for (int j = 0; j < 2; ++j) {
    const int ch  = tid + 512 * j;        // 0..1023
    const int ml  = ch >> 2, kch = ch & 3;
    const int mg  = m0 + ml;
    const int bI  = mg / HW;
    const int rem = mg - bI * HW;
    const int yy  = rem / W_;
    const int xx  = rem - yy * W_;
    aptr[j] = xcl + ((size_t)((bI * HP + yy) * WP + xx)) * C_IN + kch * 8;
    sA[j]   = ml * 40 + kch * 8;
  }
#ifndef HAVE_TDM
  const size_t bOff = (size_t)(n0 + (tid >> 2)) * C_IN + (tid & 3) * 8;
  const int    sB   = (tid >> 2) * 40 + (tid & 3) * 8;
#endif

  v8f acc[2][4] = {};
  uint4 aval[2];
#ifndef HAVE_TDM
  uint4 bval;
#endif

  auto loadA = [&](int s) {
    const int tap  = s >> 5;
    const int kc   = (s & 31) << 5;
    const int doff = ((tap / 3) * WP + (tap % 3)) * C_IN + kc;
    aval[0] = *(const uint4*)(aptr[0] + doff);
    aval[1] = *(const uint4*)(aptr[1] + doff);
  };
  auto issueB = [&](int s) {
    const int tap = s >> 5;
    const int kc  = (s & 31) << 5;
#ifdef HAVE_TDM
    if (wave == 0) {
      // D# group0: count=1 | lds_addr | global_addr(57b) | type=2
      const unsigned long long ga = (unsigned long long)(uintptr_t)
          (wp + (size_t)tap * (C_MID * C_IN) + (size_t)n0 * C_IN + kc);
      const unsigned int lds = (unsigned int)(uintptr_t)(void*)&Bs[s & 1][0];
      v4u g0 = { 1u, lds, (unsigned int)ga,
                 (unsigned int)((ga >> 32) & 0x01FFFFFFu) | 0x80000000u };
      // D# group1: data_size=2B | pad_en, interval=16dw, amount=4dw |
      //            tensor_dim0=1024, tensor_dim1=1M, tile 32x128, stride0=1024
      v8i_ g1 = { (int)0x06D10000, (int)0x04000000, 0, (int)0x00200010,
                  128, 1024, 0, 0 };
      v4i_ z4 = { 0, 0, 0, 0 };
      v8i_ z8 = { 0, 0, 0, 0, 0, 0, 0, 0 };
      __builtin_amdgcn_tensor_load_to_lds(g0, g1, z4, z4, z8, 0);
    }
#else
    bval = *(const uint4*)(wp + (size_t)tap * (C_MID * C_IN) + bOff + kc);
#endif
  };
  auto storeA = [&](int bi) {
    *(uint4*)&As[bi][sA[0]] = aval[0];
    *(uint4*)&As[bi][sA[1]] = aval[1];
#ifndef HAVE_TDM
    *(uint4*)&Bs[bi][sB] = bval;
#endif
  };
  auto compute = [&](int bi) {
    const int half = lane >> 4;
    const int row  = lane & 15;
    FragBF afr[2], bfr[4];
#pragma unroll
    for (int fm = 0; fm < 2; ++fm) {
      const unsigned short* p = &As[bi][(wm * 32 + fm * 16 + row) * 40 + half * 8];
      afr[fm].q[0] = *(const uint4*)p;
      afr[fm].q[1] = *(const uint4*)(p + 16);
    }
#pragma unroll
    for (int fn = 0; fn < 4; ++fn) {
      const unsigned short* p = &Bs[bi][(wn * 64 + fn * 16 + row) * 40 + half * 8];
      bfr[fn].q[0] = *(const uint4*)p;
      bfr[fn].q[1] = *(const uint4*)(p + 16);
    }
#pragma unroll
    for (int fm = 0; fm < 2; ++fm)
#pragma unroll
      for (int fn = 0; fn < 4; ++fn)
        acc[fm][fn] = __builtin_amdgcn_wmma_f32_16x16x32_bf16(
            false, afr[fm].v, false, bfr[fn].v, (short)0, acc[fm][fn], false, false);
  };

  const int NSTEP = 9 * 32;
  for (int s = 0; s < NSTEP; ++s) {
    issueB(s);                             // TDM (wave 0) or vector path
    loadA(s);                              // global b128 loads in flight
    if (s + 4 < NSTEP) {                   // prefetch A 4 steps ahead
      const int s4 = s + 4;
      const int t4 = s4 >> 5, k4 = (s4 & 31) << 5;
      const int d4 = ((t4 / 3) * WP + (t4 % 3)) * C_IN + k4;
      __builtin_prefetch(aptr[0] + d4, 0, 3);
      __builtin_prefetch(aptr[1] + d4, 0, 3);
    }
    if (s) compute((s - 1) & 1);           // overlap with TDM + loads
    storeA(s & 1);
#ifdef HAVE_TDM
    if (wave == 0) __builtin_amdgcn_s_wait_tensorcnt(0);
#endif
    __syncthreads();                       // publishes LDS (incl. TDM writes)
  }
  compute((NSTEP - 1) & 1);

  // epilogue: +bias, ReLU, bf16 store [M][512]
  const int half = lane >> 4, col = lane & 15;
#pragma unroll
  for (int fm = 0; fm < 2; ++fm)
#pragma unroll
    for (int fn = 0; fn < 4; ++fn) {
      const int ng = n0 + wn * 64 + fn * 16 + col;
      const float bv = bias[ng];
#pragma unroll
      for (int r = 0; r < 8; ++r) {
        const int mg = m0 + wm * 32 + fm * 16 + half * 8 + r;
        float v = acc[fm][fn][r] + bv;
        v = v > 0.f ? v : 0.f;
        act[(size_t)mg * C_MID + ng] = f2bf(v);
      }
    }
}

// ---------------- fused 1x1 heads: [M,512] x [512,64] WMMA ----------------
__global__ __launch_bounds__(256)
void head_gemm_wmma(const unsigned short* __restrict__ act,
                    const unsigned short* __restrict__ wh,
                    const float* __restrict__ bh,
                    float* __restrict__ ho) {
  __shared__ __align__(16) unsigned short As[2][128 * 40];
  __shared__ __align__(16) unsigned short Bs[2][64 * 40];
  const int tid  = threadIdx.x;
  const int lane = tid & 31;
  const int wave = tid >> 5;
  const int m0   = blockIdx.x * 128;

  v8f acc[4] = {};
  uint4 aval[2], bval;

  auto loadG = [&](int s) {
    const int kc = s << 5;
#pragma unroll
    for (int j = 0; j < 2; ++j) {
      const int ch = tid + 256 * j;
      const int mm = ch >> 2, kch = ch & 3;
      aval[j] = *(const uint4*)(act + (size_t)(m0 + mm) * C_MID + kc + kch * 8);
    }
    const int n = tid >> 2, kch = tid & 3;
    bval = *(const uint4*)(wh + (size_t)n * C_MID + kc + kch * 8);
  };
  auto storeL = [&](int bi) {
#pragma unroll
    for (int j = 0; j < 2; ++j) {
      const int ch = tid + 256 * j;
      const int mm = ch >> 2, kch = ch & 3;
      *(uint4*)&As[bi][mm * 40 + kch * 8] = aval[j];
    }
    const int n = tid >> 2, kch = tid & 3;
    *(uint4*)&Bs[bi][n * 40 + kch * 8] = bval;
  };
  auto compute = [&](int bi) {
    const int half = lane >> 4, row = lane & 15;
    FragBF af, bfr[4];
    const unsigned short* p = &As[bi][(wave * 16 + row) * 40 + half * 8];
    af.q[0] = *(const uint4*)p;
    af.q[1] = *(const uint4*)(p + 16);
#pragma unroll
    for (int fn = 0; fn < 4; ++fn) {
      const unsigned short* q = &Bs[bi][(fn * 16 + row) * 40 + half * 8];
      bfr[fn].q[0] = *(const uint4*)q;
      bfr[fn].q[1] = *(const uint4*)(q + 16);
    }
#pragma unroll
    for (int fn = 0; fn < 4; ++fn)
      acc[fn] = __builtin_amdgcn_wmma_f32_16x16x32_bf16(
          false, af.v, false, bfr[fn].v, (short)0, acc[fn], false, false);
  };

  const int NSTEP = C_MID / 32;
  for (int s = 0; s < NSTEP; ++s) {
    loadG(s);
    if (s) compute((s - 1) & 1);
    storeL(s & 1);
    __syncthreads();
  }
  compute((NSTEP - 1) & 1);

  const int half = lane >> 4, col = lane & 15;
#pragma unroll
  for (int fn = 0; fn < 4; ++fn) {
    const int ng = fn * 16 + col;
    const float bv = bh[ng];
#pragma unroll
    for (int r = 0; r < 8; ++r) {
      const int mg = m0 + wave * 16 + half * 8 + r;
      ho[(size_t)mg * NH + ng] = acc[fn][r] + bv;
    }
  }
}

// ---------------- pointwise / reductions ----------------
__global__ void softmax_prob(const float* __restrict__ ho, float* __restrict__ out) {
  int i = blockIdx.x * blockDim.x + threadIdx.x;
  if (i >= B_ * 9 * HW) return;
  int x = i % W_;
  int y = (i / W_) % H_;
  int a = (i / HW) % 9;
  int b = i / (9 * HW);
  int m = (b * H_ + y) * W_ + x;
  float s0 = ho[(size_t)m * NH + a];
  float s1 = ho[(size_t)m * NH + a + 9];
  float mx = fmaxf(s0, s1);
  float e0 = __expf(s0 - mx), e1 = __expf(s1 - mx);
  float inv = 1.f / (e0 + e1);
  out[((size_t)(b * 18 + a) * H_ + y) * W_ + x]     = e0 * inv;
  out[((size_t)(b * 18 + a + 9) * H_ + y) * W_ + x] = e1 * inv;
}

__global__ void bbox_out(const float* __restrict__ ho, float* __restrict__ out) {
  int i = blockIdx.x * blockDim.x + threadIdx.x;
  if (i >= B_ * 36 * HW) return;
  int x = i % W_;
  int y = (i / W_) % H_;
  int j = (i / HW) % 36;
  int b = i / (36 * HW);
  int m = (b * H_ + y) * W_ + x;
  out[i] = ho[(size_t)m * NH + 18 + j];
}

__global__ void cls_loss(const float* __restrict__ ho, const int* __restrict__ lab,
                         float* __restrict__ accum) {
  int i = blockIdx.x * blockDim.x + threadIdx.x;
  float v = 0.f, c = 0.f;
  if (i < B_ * 9 * HW) {
    int l = lab[i];
    if (l != -1) {
      int x = i % W_;
      int y = (i / W_) % H_;
      int a = (i / HW) % 9;
      int b = i / (9 * HW);
      int m = (b * H_ + y) * W_ + x;
      float z0 = ho[(size_t)m * NH + a];
      float z1 = ho[(size_t)m * NH + a + 9];
      float mx = fmaxf(z0, z1);
      float lse = mx + __logf(__expf(z0 - mx) + __expf(z1 - mx));
      v = lse - ((l >= 1) ? z1 : z0);
      c = 1.f;
    }
  }
  __shared__ float sv[256], sc[256];
  sv[threadIdx.x] = v; sc[threadIdx.x] = c;
  __syncthreads();
  for (int off = 128; off > 0; off >>= 1) {
    if (threadIdx.x < (unsigned)off) {
      sv[threadIdx.x] += sv[threadIdx.x + off];
      sc[threadIdx.x] += sc[threadIdx.x + off];
    }
    __syncthreads();
  }
  if (threadIdx.x == 0) { atomicAdd(&accum[0], sv[0]); atomicAdd(&accum[1], sc[0]); }
}

__global__ void box_loss(const float* __restrict__ ho, const float* __restrict__ tg,
                         const float* __restrict__ iw, const float* __restrict__ ow,
                         float* __restrict__ accum) {
  int i = blockIdx.x * blockDim.x + threadIdx.x;
  float v = 0.f;
  if (i < B_ * 36 * HW) {
    int x = i % W_;
    int y = (i / W_) % H_;
    int j = (i / HW) % 36;
    int b = i / (36 * HW);
    int m = (b * H_ + y) * W_ + x;
    float pred = ho[(size_t)m * NH + 18 + j];
    float d  = iw[i] * (pred - tg[i]);
    float ad = fabsf(d);
    float sg = (ad < (1.f / 9.f)) ? 1.f : 0.f;
    float loss = d * d * 4.5f * sg + (ad - (0.5f / 9.f)) * (1.f - sg);
    v = ow[i] * loss;
  }
  __shared__ float sv[256];
  sv[threadIdx.x] = v;
  __syncthreads();
  for (int off = 128; off > 0; off >>= 1) {
    if (threadIdx.x < (unsigned)off) sv[threadIdx.x] += sv[threadIdx.x + off];
    __syncthreads();
  }
  if (threadIdx.x == 0) atomicAdd(&accum[2], sv[0]);
}

__global__ void finalize(const float* __restrict__ accum, float* __restrict__ out) {
  if (threadIdx.x == 0) {
    out[0] = accum[0] / fmaxf(accum[1], 1.f);
    out[1] = accum[2] * (1.f / (float)B_);
  }
}

// ---------------- launcher ----------------
extern "C" void kernel_launch(void* const* d_in, const int* in_sizes, int n_in,
                              void* d_out, int out_size, void* d_ws, size_t ws_size,
                              hipStream_t stream) {
  (void)in_sizes; (void)n_in; (void)out_size; (void)ws_size;
  const float* base  = (const float*)d_in[0];
  const float* wconv = (const float*)d_in[1];
  const float* bconv = (const float*)d_in[2];
  const float* wcls  = (const float*)d_in[3];
  const float* bcls  = (const float*)d_in[4];
  const float* wbb   = (const float*)d_in[5];
  const float* bbb   = (const float*)d_in[6];
  const int*   lab   = (const int*)d_in[7];
  const float* tg    = (const float*)d_in[8];
  const float* iw    = (const float*)d_in[9];
  const float* ow    = (const float*)d_in[10];
  float* out = (float*)d_out;

  char* ws = (char*)d_ws;
  unsigned short* Wp  = (unsigned short*)(ws);             //  9,437,184 B
  unsigned short* Wh  = (unsigned short*)(ws + 9437184);   //     65,536 B
  float*          Bh  = (float*)(ws + 9502720);            //        256 B
  unsigned short* Act = (unsigned short*)(ws + 9502976);   // 25,165,824 B
  unsigned short* Xcl = (unsigned short*)(ws + 34668800);  // 54,067,200 B (dead after conv)
  float*          Ho  = (float*)(ws + 34668800);           //  6,291,456 B (aliases Xcl)
  float*          Acc = (float*)(ws + 88736000);           //         16 B

  const int XPAD4 = (B_ * HP * WP * C_IN * 2) / 16;        // 3,379,200 uint4
  zero_xpad<<<(XPAD4 + 255) / 256, 256, 0, stream>>>((uint4*)Xcl, XPAD4);
  cvt_nhwc<<<dim3(HW / 128, C_IN / 32, B_), 256, 0, stream>>>(base, Xcl);
  pack_conv_w<<<(C_MID * C_IN * 9 + 255) / 256, 256, 0, stream>>>(wconv, Wp);
  pack_heads<<<(NH * C_MID + 255) / 256, 256, 0, stream>>>(wcls, bcls, wbb, bbb, Wh, Bh);
  zero_accum<<<1, 32, 0, stream>>>(Acc);

  conv3x3_relu_wmma<<<dim3(M_TOT / 256, C_MID / 128), 512, 0, stream>>>(Xcl, Wp, bconv, Act);
  head_gemm_wmma<<<M_TOT / 128, 256, 0, stream>>>(Act, Wh, Bh, Ho);

  softmax_prob<<<(B_ * 9 * HW + 255) / 256, 256, 0, stream>>>(Ho, out);
  bbox_out<<<(B_ * 36 * HW + 255) / 256, 256, 0, stream>>>(Ho, out + OFF_BBOX);
  cls_loss<<<(B_ * 9 * HW + 255) / 256, 256, 0, stream>>>(Ho, lab, Acc);
  box_loss<<<(B_ * 36 * HW + 255) / 256, 256, 0, stream>>>(Ho, tg, iw, ow, Acc);
  finalize<<<1, 32, 0, stream>>>(Acc, out + OFF_LCLS);
}
